// VisionMambaBlock_12146167513463
// MI455X (gfx1250) — matrix operand order repaired and checked by reading, not compiled, not run on hardware
//
#include <hip/hip_runtime.h>
#include <hip/hip_bf16.h>
#include <cstdint>
#include <cstddef>

#define DIM      512
#define DT_RANK  32
#define D_STATE  16
#define BATCH    2
#define SEQ      2048
#define MROWS    (BATCH * SEQ)   /* 4096 rows of (b,l) */
#define LN_EPS   1e-5f

typedef __attribute__((ext_vector_type(16))) _Float16 v16h;
typedef __attribute__((ext_vector_type(8)))  _Float16 v8h;
typedef __attribute__((ext_vector_type(8)))  float    v8f;

// Async global->LDS copy of one 16-byte chunk (gfx1250, tracked by ASYNCcnt).
// ldsOff is the workgroup-relative LDS byte offset (low 32 bits of a generic
// pointer to __shared__); gaddr is the 64-bit global address.
__device__ __forceinline__ void async_copy_b128(uint32_t ldsOff, uint64_t gaddr) {
  asm volatile("global_load_async_to_lds_b128 %0, %1, off"
               :: "v"(ldsOff), "v"(gaddr) : "memory");
}
__device__ __forceinline__ void async_wait0() {
  asm volatile("s_wait_asynccnt 0x0" ::: "memory");
}

// ---------------------------------------------------------------- utilities

__global__ void f32_to_f16_kernel(const float* __restrict__ src,
                                  _Float16* __restrict__ dst, int n) {
  int i = blockIdx.x * blockDim.x + threadIdx.x;
  if (i < n) dst[i] = (_Float16)src[i];
}

// LayerNorm over d=512; one block (256 threads) per (b,l) row; writes f16.
__global__ void ln_kernel(const float* __restrict__ x,
                          const float* __restrict__ w,
                          const float* __restrict__ b,
                          _Float16* __restrict__ h16) {
  __shared__ float s1[256];
  __shared__ float s2[256];
  const int row = blockIdx.x;
  const int t = threadIdx.x;
  const float* xr = x + (size_t)row * DIM;
  float x0 = xr[t];
  float x1 = xr[t + 256];
  s1[t] = x0 + x1;
  s2[t] = x0 * x0 + x1 * x1;
  __syncthreads();
  for (int off = 128; off > 0; off >>= 1) {
    if (t < off) { s1[t] += s1[t + off]; s2[t] += s2[t + off]; }
    __syncthreads();
  }
  const float mean = s1[0] * (1.0f / DIM);
  const float var  = s2[0] * (1.0f / DIM) - mean * mean;
  const float rstd = rsqrtf(var + LN_EPS);
  _Float16* hr = h16 + (size_t)row * DIM;
  hr[t]       = (_Float16)((x0 - mean) * rstd * w[t]       + b[t]);
  hr[t + 256] = (_Float16)((x1 - mean) * rstd * w[t + 256] + b[t + 256]);
}

// ---------------------------------------------------------------- WMMA GEMM
// out[m,n] = act( sum_k A[m,k] * W[n,k] + bias[n] )
// A: (M x K) f16 row-major.  W: (N x K) f16 row-major (i.e. B = W^T).
// All 8 waves of a block share one 16-row weight slab W[n0:n0+16, :], which
// is staged into LDS once via async-to-LDS DMA; B fragments then come from
// ds_load_b128 while A fragments stream from global.
// act: 0 = none, 1 = silu, 2 = softplus.
__global__ __launch_bounds__(256)
void wmma_gemm_kernel(const _Float16* __restrict__ A,
                      const _Float16* __restrict__ W,
                      const float* __restrict__ bias,
                      float* __restrict__ outF,
                      _Float16* __restrict__ out16, int ncols16,
                      int N, int K, int act) {
  __shared__ __align__(32) _Float16 sW[16 * 512];   // 16 KB max (K <= 512)

  const int lane = threadIdx.x & 31;
  const int wave = threadIdx.x >> 5;
  const int m0 = (blockIdx.y * 8 + wave) * 16;
  const int n0 = blockIdx.x * 16;
  const int lh   = lane & 15;
  const int kb8  = (lane >> 4) * 8;    // A half-offset within 32-wide K tile
  const int kb16 = (lane >> 4) * 16;   // (unused)

  // ---- stage W[n0:n0+16, 0:K] into LDS (16B chunks, ASYNCcnt-tracked) ----
  const int cpr = K >> 3;              // 16-byte chunks per weight row
  const int chunks = cpr << 4;         // 16 rows
  const uint32_t ldsW = (uint32_t)(uintptr_t)(&sW[0]);
  for (int i = threadIdx.x; i < chunks; i += 256) {
    const int row = i / cpr;
    const int col = (i - row * cpr) << 3;          // half index within row
    async_copy_b128(ldsW + (uint32_t)(i << 4),
                    (uint64_t)(uintptr_t)(W + (size_t)(n0 + row) * K + col));
  }
  async_wait0();
  __syncthreads();

  const _Float16* __restrict__ arow = A + (size_t)(m0 + lh) * K;
  const _Float16* __restrict__ brow = &sW[(size_t)lh * K];

  v8f acc = {};
  for (int kt = 0; kt < K; kt += 32) {
    if (kt + 32 < K) __builtin_prefetch(arow + kt + 32 + kb8, 0, 0);
    v8h a_lo = *(const v8h*)(arow + kt + kb8);
    v8h a_hi = *(const v8h*)(arow + kt + kb8 + 16);
    v16h a = __builtin_shufflevector(a_lo, a_hi,
                                     0, 1, 2, 3, 4, 5, 6, 7,
                                     8, 9, 10, 11, 12, 13, 14, 15);
    v8h b_lo = *(const v8h*)(brow + kt + kb8);
    v8h b_hi = *(const v8h*)(brow + kt + kb8 + 16);
    v16h b = __builtin_shufflevector(b_lo, b_hi,
                                     0, 1, 2, 3, 4, 5, 6, 7,
                                     8, 9, 10, 11, 12, 13, 14, 15);     // ds_load_b128 x2
    acc = __builtin_amdgcn_wmma_f32_16x16x32_f16(
        /*neg_a=*/false, a, /*neg_b=*/false, b,
        /*c_mod=*/(short)0, acc, /*reuse_a=*/false, /*reuse_b=*/false);
  }

  const int nn = n0 + lh;
  const float bv = bias ? bias[nn] : 0.0f;
  const int mbase = m0 + (lane >> 4) * 8;
#pragma unroll
  for (int r = 0; r < 8; ++r) {
    float v = acc[r] + bv;
    if (act == 1) v = v / (1.0f + __expf(-v));                     // silu
    else if (act == 2) v = (v > 20.0f) ? v : log1pf(__expf(v));    // softplus
    const int mm = mbase + r;
    if (outF)  outF[(size_t)mm * N + nn] = v;
    if (out16 && nn < ncols16) out16[(size_t)mm * ncols16 + nn] = (_Float16)v;
  }
}

// ---------------------------------------------------------------- scan
// Per-(batch, channel) sequential selective scan, forward (dir=+1) or
// reverse (dir=-1) in time. B/C (shared across channels) staged into LDS via
// async-to-LDS DMA in 64-step windows: chunk j covers row tt=j>>2, four
// floats q=j&3 -> LDS byte offset j*16, one B-chunk + one C-chunk per thread.
__global__ __launch_bounds__(256)
void scan_kernel(const float* __restrict__ xin,    // (b,l,d) branch input
                 const float* __restrict__ delta,  // (b,l,d)
                 const float* __restrict__ dbc,    // (b,l,64): [32:48]=B, [48:64]=C
                 const float* __restrict__ Alog,   // (d,16)
                 const float* __restrict__ Dp,     // (d,)
                 float* __restrict__ y,            // (b,l,d)
                 int dir) {
  __shared__ __align__(16) float sB[64 * D_STATE];
  __shared__ __align__(16) float sC[64 * D_STATE];
  const int dch = blockIdx.x * blockDim.x + threadIdx.x;  // channel
  const int b = blockIdx.y;

  float a[D_STATE];
#pragma unroll
  for (int n = 0; n < D_STATE; ++n) a[n] = -__expf(Alog[(size_t)dch * D_STATE + n]);
  float s[D_STATE];
#pragma unroll
  for (int n = 0; n < D_STATE; ++n) s[n] = 0.0f;
  const float Dv = Dp[dch];

  const uint32_t ldsB = (uint32_t)(uintptr_t)(&sB[0]);
  const uint32_t ldsC = (uint32_t)(uintptr_t)(&sC[0]);

  for (int chunk = 0; chunk < SEQ; chunk += 64) {
    __syncthreads();
    {
      const int j = threadIdx.x;                 // 0..255: one 16B chunk each
      const int tt = j >> 2, q = j & 3;
      const int t = chunk + tt;
      const int l = (dir > 0) ? t : (SEQ - 1 - t);
      const float* rowp = dbc + ((size_t)b * SEQ + l) * 64;
      async_copy_b128(ldsB + (uint32_t)(j << 4),
                      (uint64_t)(uintptr_t)(rowp + DT_RANK + q * 4));
      async_copy_b128(ldsC + (uint32_t)(j << 4),
                      (uint64_t)(uintptr_t)(rowp + DT_RANK + D_STATE + q * 4));
    }
    async_wait0();
    __syncthreads();
    for (int tt = 0; tt < 64; ++tt) {
      const int t = chunk + tt;
      const int l = (dir > 0) ? t : (SEQ - 1 - t);
      const size_t idx = ((size_t)b * SEQ + l) * DIM + dch;
      const float dv = delta[idx];
      const float xv = xin[idx];
      const float dx = dv * xv;
      float yv = 0.0f;
#pragma unroll
      for (int n = 0; n < D_STATE; ++n) {
        const float dA = __expf(dv * a[n]);
        s[n] = dA * s[n] + dx * sB[tt * D_STATE + n];
        yv += s[n] * sC[tt * D_STATE + n];
      }
      y[idx] = yv + Dv * xv;
    }
  }
}

// ---------------------------------------------------------------- combine
__global__ void combine_kernel(const float* __restrict__ y1,
                               const float* __restrict__ y2,
                               const float* __restrict__ z,
                               const float* __restrict__ skip,
                               float* __restrict__ out, int n) {
  int i = blockIdx.x * blockDim.x + threadIdx.x;
  if (i < n) out[i] = (y1[i] + y2[i]) * z[i] + skip[i];
}

// ---------------------------------------------------------------- launch

extern "C" void kernel_launch(void* const* d_in, const int* in_sizes, int n_in,
                              void* d_out, int out_size, void* d_ws, size_t ws_size,
                              hipStream_t stream) {
  const float* x       = (const float*)d_in[0];
  const float* ln_w    = (const float*)d_in[1];
  const float* ln_b    = (const float*)d_in[2];
  const float* Wz      = (const float*)d_in[3];
  const float* bz      = (const float*)d_in[4];
  const float* Wx      = (const float*)d_in[5];
  const float* bx      = (const float*)d_in[6];
  const float* Wf      = (const float*)d_in[7];
  const float* bf      = (const float*)d_in[8];
  const float* Wb      = (const float*)d_in[9];
  const float* bb      = (const float*)d_in[10];
  const float* f_Wdbc  = (const float*)d_in[11];
  const float* f_Wdt   = (const float*)d_in[12];
  const float* f_bdt   = (const float*)d_in[13];
  const float* f_Alog  = (const float*)d_in[14];
  const float* f_D     = (const float*)d_in[15];
  const float* b_Wdbc  = (const float*)d_in[16];
  const float* b_Wdt   = (const float*)d_in[17];
  const float* b_bdt   = (const float*)d_in[18];
  const float* b_Alog  = (const float*)d_in[19];
  const float* b_D     = (const float*)d_in[20];
  float* out = (float*)d_out;

  // ---- workspace carve-up (all offsets 256B-aligned) ----
  char* ws = (char*)d_ws;
  size_t off = 0;
  auto alloc = [&](size_t bytes) -> char* {
    char* p = ws + off;
    off += (bytes + 255) & ~(size_t)255;
    return p;
  };
  const size_t MD16 = (size_t)MROWS * DIM * sizeof(_Float16);
  const size_t MD32 = (size_t)MROWS * DIM * sizeof(float);

  _Float16* h16     = (_Float16*)alloc(MD16);
  _Float16* Wz16    = (_Float16*)alloc((size_t)DIM * DIM * 2);
  _Float16* Wx16    = (_Float16*)alloc((size_t)DIM * DIM * 2);
  _Float16* Wf16    = (_Float16*)alloc((size_t)DIM * DIM * 2);
  _Float16* Wb16    = (_Float16*)alloc((size_t)DIM * DIM * 2);
  _Float16* fWdbc16 = (_Float16*)alloc((size_t)64 * DIM * 2);
  _Float16* bWdbc16 = (_Float16*)alloc((size_t)64 * DIM * 2);
  _Float16* fWdt16  = (_Float16*)alloc((size_t)DIM * DT_RANK * 2);
  _Float16* bWdt16  = (_Float16*)alloc((size_t)DIM * DT_RANK * 2);
  float*    zbuf    = (float*)alloc(MD32);
  _Float16* xp16    = (_Float16*)alloc(MD16);
  float*    x1      = (float*)alloc(MD32);
  _Float16* x116    = (_Float16*)alloc(MD16);
  float*    xb      = (float*)alloc(MD32);
  _Float16* xb16    = (_Float16*)alloc(MD16);
  float*    dbc_f   = (float*)alloc((size_t)MROWS * 64 * 4);
  float*    dbc_b   = (float*)alloc((size_t)MROWS * 64 * 4);
  _Float16* ddt_f16 = (_Float16*)alloc((size_t)MROWS * DT_RANK * 2);
  _Float16* ddt_b16 = (_Float16*)alloc((size_t)MROWS * DT_RANK * 2);
  float*    delta_f = (float*)alloc(MD32);
  float*    delta_b = (float*)alloc(MD32);
  float*    y1      = (float*)alloc(MD32);
  float*    y2      = (float*)alloc(MD32);
  if (off > ws_size) return;  // workspace too small; nothing safe to do

  // ---- weight conversion to f16 ----
  auto cvt = [&](const float* s, _Float16* d, int n) {
    f32_to_f16_kernel<<<(n + 255) / 256, 256, 0, stream>>>(s, d, n);
  };
  cvt(Wz, Wz16, DIM * DIM);
  cvt(Wx, Wx16, DIM * DIM);
  cvt(Wf, Wf16, DIM * DIM);
  cvt(Wb, Wb16, DIM * DIM);
  cvt(f_Wdbc, fWdbc16, 64 * DIM);
  cvt(b_Wdbc, bWdbc16, 64 * DIM);
  cvt(f_Wdt, fWdt16, DIM * DT_RANK);
  cvt(b_Wdt, bWdt16, DIM * DT_RANK);

  // ---- LayerNorm ----
  ln_kernel<<<MROWS, 256, 0, stream>>>(x, ln_w, ln_b, h16);

  const dim3 blk(256);
  auto gemm_grid = [](int N) { return dim3(N / 16, MROWS / 128); };

  // z = silu(h @ Wz^T + bz)                       (f32 only)
  wmma_gemm_kernel<<<gemm_grid(DIM), blk, 0, stream>>>(
      h16, Wz16, bz, zbuf, nullptr, 0, DIM, DIM, /*act=*/1);
  // xp = h @ Wx^T + bx                            (f16 only; feeds next GEMMs)
  wmma_gemm_kernel<<<gemm_grid(DIM), blk, 0, stream>>>(
      h16, Wx16, bx, nullptr, xp16, DIM, DIM, DIM, /*act=*/0);
  // x1 = softplus(xp @ Wf^T + bf)                 (f32 + f16)
  wmma_gemm_kernel<<<gemm_grid(DIM), blk, 0, stream>>>(
      xp16, Wf16, bf, x1, x116, DIM, DIM, DIM, /*act=*/2);
  // xb = softplus(xp @ Wb^T + bb)  (flip is pointwise-invariant; scan reverses)
  wmma_gemm_kernel<<<gemm_grid(DIM), blk, 0, stream>>>(
      xp16, Wb16, bb, xb, xb16, DIM, DIM, DIM, /*act=*/2);
  // dbc_f = x1 @ f_Wdbc^T   (N=64; f16 copy of first 32 cols for delta GEMM)
  wmma_gemm_kernel<<<gemm_grid(64), blk, 0, stream>>>(
      x116, fWdbc16, nullptr, dbc_f, ddt_f16, DT_RANK, 64, DIM, /*act=*/0);
  wmma_gemm_kernel<<<gemm_grid(64), blk, 0, stream>>>(
      xb16, bWdbc16, nullptr, dbc_b, ddt_b16, DT_RANK, 64, DIM, /*act=*/0);
  // delta = softplus(dbc[:, :32] @ Wdt^T + bdt)   (K=32 -> single WMMA step)
  wmma_gemm_kernel<<<gemm_grid(DIM), blk, 0, stream>>>(
      ddt_f16, fWdt16, f_bdt, delta_f, nullptr, 0, DIM, DT_RANK, /*act=*/2);
  wmma_gemm_kernel<<<gemm_grid(DIM), blk, 0, stream>>>(
      ddt_b16, bWdt16, b_bdt, delta_b, nullptr, 0, DIM, DT_RANK, /*act=*/2);

  // ---- selective scans (fwd / reverse-time) ----
  const dim3 sgrid(DIM / 256, BATCH);
  scan_kernel<<<sgrid, 256, 0, stream>>>(x1, delta_f, dbc_f, f_Alog, f_D, y1, +1);
  scan_kernel<<<sgrid, 256, 0, stream>>>(xb, delta_b, dbc_b, b_Alog, b_D, y2, -1);

  // ---- out = (y1 + y2) * z + skip ----
  const int total = MROWS * DIM;
  combine_kernel<<<(total + 255) / 256, 256, 0, stream>>>(y1, y2, zbuf, x, out, total);
}
